// SeqNet_45114336477966
// MI455X (gfx1250) — matrix operand order repaired and checked by reading, not compiled, not run on hardware
//
#include <hip/hip_runtime.h>
#include <hip/hip_bf16.h>
#include <stdint.h>

// ---------------------------------------------------------------------------
// 3-layer LSTM (B=64, T=512, H=256, C=1024), second weight set only
// (the reference discards the first lstm3's result).
// bf16 WMMA (v_wmma_f32_16x16x32_bf16); fp32 gate math.
// GEMM: 16x64/wave with explicit 2-stage software pipeline (ping-pong regs).
// Recurrence: W_hh fully register-resident; xg loads overlapped with WMMAs.
// ---------------------------------------------------------------------------

typedef __bf16 bf16;
typedef __attribute__((ext_vector_type(8)))  __bf16 v8bf;
typedef __attribute__((ext_vector_type(16))) __bf16 v16bf;
typedef __attribute__((ext_vector_type(8)))  float  v8f;

#define HID   256
#define FOURH 1024
#define TT    512
#define BB    64
#define RR    (BB * TT)     // 32768 rows of (b,t)

// ---- operand loaders matching CDNA5 WMMA VGPR layouts (ISA 7.12.2) --------
__device__ __forceinline__ v16bf load_a_tile(const bf16* row_base, int k0, int half) {
    const v8bf lo = *(const v8bf*)(row_base + k0 + half * 8);
    const v8bf hi = *(const v8bf*)(row_base + k0 + 16 + half * 8);
    v16bf a;
#pragma unroll
    for (int i = 0; i < 8; ++i) { a[i] = lo[i]; a[i + 8] = hi[i]; }
    return a;
}
__device__ __forceinline__ v16bf load_b_tile(const bf16* col_base, int k0, int half) {
    const bf16* p = col_base + k0 + half * 8;
    const v8bf lo = *(const v8bf*)(p);
    const v8bf hi = *(const v8bf*)(p + 16);
    v16bf b;
#pragma unroll
    for (int i = 0; i < 8; ++i) { b[i] = lo[i]; b[i + 8] = hi[i]; }
    return b;
}

__device__ __forceinline__ float sigm(float x) { return 1.0f / (1.0f + __expf(-x)); }

// ---------------------------------------------------------------------------
__global__ void f2bf_kernel(const float* __restrict__ in, bf16* __restrict__ out, int n) {
    int i = blockIdx.x * blockDim.x + threadIdx.x;
    if (i < n) out[i] = (bf16)in[i];
}

// ---------------------------------------------------------------------------
// xg = A[R,K](bf16) * W[N,K]^T (bf16) + bias[N]  ->  out[R,N] f32
// 16x64 per wave; software-pipelined: loads for k+32 overlap WMMAs for k.
// K must be a multiple of 64 (holds: 1024 and 256).
// ---------------------------------------------------------------------------
__global__ __launch_bounds__(256) void gemm_xg_kernel(
    const bf16* __restrict__ A, int lda,
    const bf16* __restrict__ W,
    const float* __restrict__ bias,
    float* __restrict__ out, int R, int K, int N)
{
    const int wave = (blockIdx.x * blockDim.x + threadIdx.x) >> 5;
    const int lane = threadIdx.x & 31;
    const int nblk = N >> 6;
    const int mt   = wave / nblk;
    const int nb   = wave % nblk;
    if (mt * 16 >= R) return;                 // wave-uniform; EXEC stays all-ones

    const int half = lane >> 4;
    const int l15  = lane & 15;
    const bf16* arow = A + (size_t)(mt * 16 + l15) * lda;
    const bf16* bcol[4];
#pragma unroll
    for (int j = 0; j < 4; ++j)
        bcol[j] = W + (size_t)(nb * 64 + j * 16 + l15) * K;

    v8f acc[4] = {};

    // prologue: fill buffer 0 for k=0
    v16bf aA0 = load_a_tile(arow, 0, half);
    v16bf bB0[4];
#pragma unroll
    for (int j = 0; j < 4; ++j) bB0[j] = load_b_tile(bcol[j], 0, half);

    for (int k0 = 0; k0 < K; k0 += 64) {
        // prefetch buffer 1 for k0+32 while consuming buffer 0
        v16bf aA1 = load_a_tile(arow, k0 + 32, half);
        v16bf bB1[4];
#pragma unroll
        for (int j = 0; j < 4; ++j) bB1[j] = load_b_tile(bcol[j], k0 + 32, half);
#pragma unroll
        for (int j = 0; j < 4; ++j)
            acc[j] = __builtin_amdgcn_wmma_f32_16x16x32_bf16(
                false, aA0, false, bB0[j], (short)0, acc[j], false, false);

        // prefetch buffer 0 for k0+64 (dummy k=0 on last iter) while consuming buffer 1
        const int kn = (k0 + 64 < K) ? (k0 + 64) : 0;
        aA0 = load_a_tile(arow, kn, half);
#pragma unroll
        for (int j = 0; j < 4; ++j) bB0[j] = load_b_tile(bcol[j], kn, half);
#pragma unroll
        for (int j = 0; j < 4; ++j)
            acc[j] = __builtin_amdgcn_wmma_f32_16x16x32_bf16(
                false, aA1, false, bB1[j], (short)0, acc[j], false, false);
    }

#pragma unroll
    for (int j = 0; j < 4; ++j) {
        const int n  = nb * 64 + j * 16 + l15;
        const float bv = bias[n];
        float* obase = out + (size_t)(mt * 16) * N + n;
#pragma unroll
        for (int r = 0; r < 8; ++r) {
            const int mm = r + half * 8;
            obase[(size_t)mm * N] = acc[j][r] + bv;
        }
    }
}

// ---------------------------------------------------------------------------
// LSTM recurrence, one layer. 4 blocks x 16 batch rows, 16 waves each.
// Wave w owns hidden cols [16w,16w+16) for ALL 4 gates -> gate math + cell
// state in registers. Entire W_hh slice register-resident (256 VGPRs/lane).
// WMMA chains start at 0; xg added in the activation phase so the 32 xg
// loads issued at step start are hidden behind the 32 WMMAs.
// ---------------------------------------------------------------------------
__global__ __launch_bounds__(512) void lstm_rec_kernel(
    const float* __restrict__ xg,       // [B,T,4H] f32 (bias already folded in)
    const bf16*  __restrict__ whh,      // [4H,H] bf16
    bf16*        __restrict__ hseq,     // [B,T,H] bf16 out
    int T)
{
    __shared__ bf16 sh[16 * HID];       // 8 KB hidden state (WMMA A operand)

    const int tid  = threadIdx.x;
    const int lane = tid & 31;
    const int wave = tid >> 5;
    const int half = lane >> 4;
    const int l15  = lane & 15;
    const int b0   = blockIdx.x * 16;
    const int ncol = wave * 16 + l15;   // hidden column owned by this lane

    // preload the full W_hh slice for this wave: 4 gates x 8 k-steps
    v16bf wB[4][8];
#pragma unroll
    for (int g = 0; g < 4; ++g) {
        const bf16* bcolg = whh + (size_t)(g * 256 + ncol) * HID;
#pragma unroll
        for (int k = 0; k < 8; ++k)
            wB[g][k] = load_b_tile(bcolg, k * 32, half);
    }

    for (int i = tid; i < 16 * HID; i += 512) sh[i] = (bf16)0.0f;
    v8f creg = {};                      // cell state: this lane's 8 (m,n) elems
    __syncthreads();

    const bf16* arow = sh + l15 * HID;

    for (int t = 0; t < T; ++t) {
        // ---- issue xg loads (consumed only after all WMMAs) ----
        v8f xv[4];
#pragma unroll
        for (int g = 0; g < 4; ++g) {
#pragma unroll
            for (int r = 0; r < 8; ++r) {
                const int mm = r + half * 8;
                xv[g][r] = xg[(((size_t)(b0 + mm)) * T + t) * FOURH + g * 256 + ncol];
            }
        }
        // ---- gates = h @ Whh^T : register B operands, 4 WMMA chains ----
        v8f acc[4] = {};
#pragma unroll
        for (int k0 = 0; k0 < 8; ++k0) {
            const v16bf a = load_a_tile(arow, k0 * 32, half);
#pragma unroll
            for (int g = 0; g < 4; ++g)
                acc[g] = __builtin_amdgcn_wmma_f32_16x16x32_bf16(
                    false, a, false, wB[g][k0], (short)0, acc[g], false, false);
        }
        __syncthreads();                // everyone done reading sh for step t

        // ---- LSTM nonlinearity fully in registers (i,f,g,o order) ----
#pragma unroll
        for (int r = 0; r < 8; ++r) {
            const int mm = r + half * 8;
            const float gi = acc[0][r] + xv[0][r];
            const float gf = acc[1][r] + xv[1][r];
            const float gg = acc[2][r] + xv[2][r];
            const float go = acc[3][r] + xv[3][r];
            const float c  = sigm(gf) * creg[r] + sigm(gi) * tanhf(gg);
            const float h  = sigm(go) * tanhf(c);
            creg[r] = c;
            const bf16 hb = (bf16)h;
            sh[mm * HID + ncol] = hb;
            hseq[(((size_t)(b0 + mm)) * T + t) * HID + ncol] = hb;
        }
        __syncthreads();                // new h visible to all waves
    }
}

// ---------------------------------------------------------------------------
__global__ __launch_bounds__(256) void rowsum_kernel(
    const bf16* __restrict__ h, float* __restrict__ out, int rows)
{
    const int gw   = (blockIdx.x * blockDim.x + threadIdx.x) >> 5;
    const int lane = threadIdx.x & 31;
    if (gw >= rows) return;
    const v8bf v = *(const v8bf*)(h + (size_t)gw * HID + lane * 8);
    float s = 0.0f;
#pragma unroll
    for (int i = 0; i < 8; ++i) s += (float)v[i];
#pragma unroll
    for (int off = 16; off > 0; off >>= 1) s += __shfl_down(s, off, 32);
    if (lane == 0) out[gw] = s;
}

// ---------------------------------------------------------------------------
extern "C" void kernel_launch(void* const* d_in, const int* in_sizes, int n_in,
                              void* d_out, int out_size, void* d_ws, size_t ws_size,
                              hipStream_t stream) {
    const float* x      = (const float*)d_in[0];  // [64,512,1024]
    const float* w_ih0  = (const float*)d_in[1];  // [2,1024,1024]
    const float* w_ih12 = (const float*)d_in[2];  // [2,2,1024,256]
    const float* w_hh   = (const float*)d_in[3];  // [2,3,1024,256]
    const float* bvec   = (const float*)d_in[4];  // [2,3,1024]
    float* out = (float*)d_out;                   // [64,512]

    char* ws = (char*)d_ws;
    float* xgbuf  = (float*)(ws + 0);                        // 128 MiB
    bf16*  xbf    = (bf16*) (ws + (size_t)134217728);        //  64 MiB
    bf16*  hseq0  = (bf16*) (ws + (size_t)201326592);        //  16 MiB
    bf16*  hseq1  = (bf16*) (ws + (size_t)218103808);        //  16 MiB
    bf16*  wih0bf = (bf16*) (ws + (size_t)234881024);        //   2 MiB
    bf16*  wih1bf = (bf16*) (ws + (size_t)236978176);
    bf16*  wih2bf = (bf16*) (ws + (size_t)237502464);
    bf16*  whh0bf = (bf16*) (ws + (size_t)238026752);
    bf16*  whh1bf = (bf16*) (ws + (size_t)238551040);
    bf16*  whh2bf = (bf16*) (ws + (size_t)239075328);

    const int NX   = RR * 1024;
    const int NW0  = 1024 * 1024;
    const int NW   = 1024 * 256;

    f2bf_kernel<<<(NX  + 255) / 256, 256, 0, stream>>>(x, xbf, NX);
    f2bf_kernel<<<(NW0 + 255) / 256, 256, 0, stream>>>(w_ih0  + (size_t)1 * NW0, wih0bf, NW0);
    f2bf_kernel<<<(NW  + 255) / 256, 256, 0, stream>>>(w_ih12 + (size_t)2 * NW,  wih1bf, NW);
    f2bf_kernel<<<(NW  + 255) / 256, 256, 0, stream>>>(w_ih12 + (size_t)3 * NW,  wih2bf, NW);
    f2bf_kernel<<<(NW  + 255) / 256, 256, 0, stream>>>(w_hh   + (size_t)3 * NW,  whh0bf, NW);
    f2bf_kernel<<<(NW  + 255) / 256, 256, 0, stream>>>(w_hh   + (size_t)4 * NW,  whh1bf, NW);
    f2bf_kernel<<<(NW  + 255) / 256, 256, 0, stream>>>(w_hh   + (size_t)5 * NW,  whh2bf, NW);

    const float* bias0 = bvec + (size_t)3 * 1024;
    const float* bias1 = bvec + (size_t)4 * 1024;
    const float* bias2 = bvec + (size_t)5 * 1024;

    const int waves   = (RR / 16) * (FOURH / 64);   // 32768 waves
    const int gblocks = waves / 8;

    gemm_xg_kernel<<<gblocks, 256, 0, stream>>>(xbf, 1024, wih0bf, bias0, xgbuf, RR, 1024, FOURH);
    lstm_rec_kernel<<<4, 512, 0, stream>>>(xgbuf, whh0bf, hseq0, TT);

    gemm_xg_kernel<<<gblocks, 256, 0, stream>>>(hseq0, HID, wih1bf, bias1, xgbuf, RR, HID, FOURH);
    lstm_rec_kernel<<<4, 512, 0, stream>>>(xgbuf, whh1bf, hseq1, TT);

    gemm_xg_kernel<<<gblocks, 256, 0, stream>>>(hseq1, HID, wih2bf, bias2, xgbuf, RR, HID, FOURH);
    lstm_rec_kernel<<<4, 512, 0, stream>>>(xgbuf, whh2bf, hseq0, TT);

    rowsum_kernel<<<(RR * 32 + 255) / 256, 256, 0, stream>>>(hseq0, out, RR);
}
